// Discriminator_58136677319040
// MI455X (gfx1250) — matrix operand order repaired and checked by reading, not compiled, not run on hardware
//
#include <hip/hip_runtime.h>

#define B_  16
#define N_  2048
#define NH_ 64
#define K1_ 1024
#define K2_ 512

typedef __attribute__((ext_vector_type(16))) _Float16 v16h;
typedef __attribute__((ext_vector_type(8)))  _Float16 v8h;
typedef __attribute__((ext_vector_type(8)))  float    v8f;

__device__ __forceinline__ float sigm(float x) { return 1.0f / (1.0f + __expf(-x)); }

__device__ __forceinline__ unsigned hash2(unsigned n, unsigned h, unsigned seed) {
    unsigned x = n * 0x9E3779B1u ^ (h + 0x7F4A7C15u) * 0x85EBCA77u ^ (seed + 1u) * 0xC2B2AE3Du;
    x ^= x >> 16; x *= 0x7FEB352Du; x ^= x >> 15; x *= 0x846CA68Bu; x ^= x >> 16;
    return x;
}
#define DROP_THRESH 429496729u  // 0.1 * 2^32

__device__ __forceinline__ void wave_reduce4(float& a, float& b, float& c, float& d) {
    for (int off = 16; off > 0; off >>= 1) {
        a += __shfl_xor(a, off, 32);
        b += __shfl_xor(b, off, 32);
        c += __shfl_xor(c, off, 32);
        d += __shfl_xor(d, off, 32);
    }
}

// ---------------- 1) dropout + fusion + f16-transposed copies ----------------
__global__ void prep_kernel(const float* __restrict__ pos1, const float* __restrict__ pos2,
                            const float* __restrict__ neg1, const float* __restrict__ neg2,
                            float* __restrict__ p1, float* __restrict__ p2,
                            float* __restrict__ fpos, float* __restrict__ fneg,
                            _Float16* __restrict__ p1T, _Float16* __restrict__ p2T) {
    size_t t = (size_t)blockIdx.x * 256 + threadIdx.x;     // < B*N*NH
    int h = (int)(t & 63);
    size_t bn = t >> 6;
    int n = (int)(bn & (N_ - 1));
    int b = (int)(bn >> 11);
    bool m1 = hash2(n, h, 0) < DROP_THRESH;
    bool m2 = hash2(n, h, 1) < DROP_THRESH;
    float p1v = m1 ? 0.0f : pos1[t];
    float n1v = m1 ? 0.0f : neg1[t];
    float p2v = m2 ? 0.0f : pos2[t];
    float n2v = m2 ? 0.0f : neg2[t];
    p1[t] = p1v; p2[t] = p2v;
    fpos[t] = 0.5f * (p1v + p2v);
    fneg[t] = 0.5f * (n1v + n2v);
    size_t ti = ((size_t)b * NH_ + h) * N_ + n;            // [B, NH, N] transposed
    p1T[ti] = (_Float16)p1v;
    p2T[ti] = (_Float16)p2v;
}

// ---------------- 2) readout: mean over nodes -> sigmoid ----------------
__global__ void mean_sig_kernel(const float* __restrict__ p1, const float* __restrict__ p2,
                                float* __restrict__ g1, float* __restrict__ g2) {
    int bh = blockIdx.x;                  // b*64 + h
    int b = bh >> 6, h = bh & 63;
    __shared__ float s1[256], s2[256];
    float a1 = 0.f, a2 = 0.f;
    for (int n = threadIdx.x; n < N_; n += 256) {
        size_t i = ((size_t)b * N_ + n) * NH_ + h;
        a1 += p1[i]; a2 += p2[i];
    }
    s1[threadIdx.x] = a1; s2[threadIdx.x] = a2;
    __syncthreads();
    for (int off = 128; off > 0; off >>= 1) {
        if ((int)threadIdx.x < off) {
            s1[threadIdx.x] += s1[threadIdx.x + off];
            s2[threadIdx.x] += s2[threadIdx.x + off];
        }
        __syncthreads();
    }
    if (threadIdx.x == 0) {
        g1[bh] = sigm(s1[0] * (1.0f / N_));
        g2[bh] = sigm(s2[0] * (1.0f / N_));
    }
}

// ---------------- 3) Wg = W0 @ g (bilinear vs broadcast vector) ----------------
__global__ void wg_kernel(const float* __restrict__ W0, const float* __restrict__ g1,
                          const float* __restrict__ g2, float* __restrict__ Wg1,
                          float* __restrict__ Wg2) {
    int b = blockIdx.x, i = threadIdx.x;  // 64 threads
    float a1 = 0.f, a2 = 0.f;
    for (int j = 0; j < NH_; ++j) {
        float w = W0[i * NH_ + j];
        a1 += w * g1[b * NH_ + j];
        a2 += w * g2[b * NH_ + j];
    }
    Wg1[b * NH_ + i] = a1;
    Wg2[b * NH_ + i] = a2;
}

// ---------------- 4) stage-0 logits: wave per row ----------------
__global__ void logits0_kernel(const float* __restrict__ fpos, const float* __restrict__ fneg,
                               const float* __restrict__ Wg1, const float* __restrict__ Wg2,
                               const float* __restrict__ b0, const float* __restrict__ alpha,
                               float* __restrict__ out0) {
    int gw = (int)((blockIdx.x * blockDim.x + threadIdx.x) >> 5);  // row id in [0, B*N)
    int lane = threadIdx.x & 31;
    int b = gw >> 11, n = gw & (N_ - 1);
    size_t base = (size_t)gw * NH_;
    float x0p = fpos[base + lane], x1p = fpos[base + lane + 32];
    float x0n = fneg[base + lane], x1n = fneg[base + lane + 32];
    float w10 = Wg1[b * NH_ + lane], w11 = Wg1[b * NH_ + lane + 32];
    float w20 = Wg2[b * NH_ + lane], w21 = Wg2[b * NH_ + lane + 32];
    float d1p = x0p * w10 + x1p * w11;
    float d2p = x0p * w20 + x1p * w21;
    float d1n = x0n * w10 + x1n * w11;
    float d2n = x0n * w20 + x1n * w21;
    wave_reduce4(d1p, d2p, d1n, d2n);
    if (lane == 0) {
        float bb = b0[0], al = alpha[0];
        out0[(size_t)b * (2 * N_) + n]      = al * (d1p + bb) + (1.0f - al) * (d2p + bb);
        out0[(size_t)b * (2 * N_) + N_ + n] = al * (d1n + bb) + (1.0f - al) * (d2n + bb);
    }
}

// ---------------- 5) top-k via in-LDS bitonic sort (descending) ----------------
extern __shared__ unsigned char smem_dyn[];
__global__ void topk_kernel(const float* __restrict__ logits, int M, int K,
                            int* __restrict__ idx_out) {
    float* s = (float*)smem_dyn;
    int* id = (int*)(s + M);
    int b = blockIdx.x;
    const float* L = logits + (size_t)b * (2 * M);
    for (int i = threadIdx.x; i < M; i += blockDim.x) {
        s[i] = sigm(L[M + i] - L[i]);   // neg - pos
        id[i] = i;
    }
    __syncthreads();
    for (int k = 2; k <= M; k <<= 1) {
        for (int j = k >> 1; j > 0; j >>= 1) {
            for (int t = threadIdx.x; t < (M >> 1); t += blockDim.x) {
                int i = 2 * t - (t & (j - 1));
                int p = i + j;
                bool dirDesc = ((i & k) == 0);
                float a = s[i], c = s[p];
                bool agt = (a > c) || (a == c && id[i] < id[p]);
                if (dirDesc ? !agt : agt) {
                    s[i] = c; s[p] = a;
                    int ti = id[i]; id[i] = id[p]; id[p] = ti;
                }
            }
            __syncthreads();
        }
    }
    for (int t = threadIdx.x; t < K; t += blockDim.x) idx_out[b * K + t] = id[t];
}

// ---------------- 6) jnp.sort(idx, axis=0): per-column sort across batch ----------------
__global__ void colsort_kernel(int* __restrict__ idx, int K) {
    int j = blockIdx.x * blockDim.x + threadIdx.x;
    if (j >= K) return;
    int v[B_];
    for (int b = 0; b < B_; ++b) v[b] = idx[b * K + j];
    for (int a = 1; a < B_; ++a) {
        int key = v[a], c = a - 1;
        while (c >= 0 && v[c] > key) { v[c + 1] = v[c]; --c; }
        v[c + 1] = key;
    }
    for (int b = 0; b < B_; ++b) idx[b * K + j] = v[b];
}

// ---------------- 7) gather fus arrays by idx ----------------
__global__ void gather1_kernel(const float* __restrict__ fpos, const float* __restrict__ fneg,
                               const int* __restrict__ idx, float* __restrict__ selp,
                               float* __restrict__ seln) {
    size_t t = (size_t)blockIdx.x * 256 + threadIdx.x;     // B*K1*NH
    int h = (int)(t & 63);
    size_t bj = t >> 6;
    int j = (int)(bj & (K1_ - 1));
    int b = (int)(bj >> 10);
    int r = idx[b * K1_ + j];
    size_t src = ((size_t)b * N_ + r) * NH_ + h;
    selp[t] = fpos[src];
    seln[t] = fneg[src];
}

// ---------------- 8) heavy GEMM on selected rows, WMMA f16 ----------------
// One wave owns a full 16x64 output tile: A fragment loaded straight from
// global (no LDS, no barriers), 4 WMMAs per 32-wide K step.
__global__ __launch_bounds__(256) void gemm_sigmoid_wmma(
    const float* __restrict__ adj, const float* __restrict__ diff,
    const _Float16* __restrict__ p1T, const _Float16* __restrict__ p2T,
    const int* __restrict__ idx, float* __restrict__ l1, float* __restrict__ l2) {
    const int b = blockIdx.y;
    const int z = blockIdx.z;                        // 0: adj/p1, 1: diff/p2
    const float* A = z ? diff : adj;
    const _Float16* Bt = z ? p2T : p1T;
    float* Out = z ? l2 : l1;

    const int wave = threadIdx.x >> 5;
    const int lane = threadIdx.x & 31;
    const int rt = blockIdx.x * 8 + wave;            // row tile 0..K1/16-1
    const int m15 = lane & 15;
    const int hi = lane >> 4;                        // 0 or 1

    // Each lane owns one gathered A row (lanes 16-31 mirror rows of 0-15).
    const int row = idx[b * K1_ + rt * 16 + m15];
    const float* arow = A + ((size_t)b * N_ + row) * N_;
    const float* ag0 = arow + hi * 16;               // K group 0: hi*16 .. hi*16+7
    const float* ag1 = arow + hi * 16 + 8;           // K group 1: hi*16+8 .. +7

    // B fragments: lane owns NH column c*16+m15; contiguous K halves (transposed).
    const _Float16* bp0 = Bt + ((size_t)b * NH_ +  0 + m15) * N_ + hi * 16;
    const _Float16* bp1 = Bt + ((size_t)b * NH_ + 16 + m15) * N_ + hi * 16;
    const _Float16* bp2 = Bt + ((size_t)b * NH_ + 32 + m15) * N_ + hi * 16;
    const _Float16* bp3 = Bt + ((size_t)b * NH_ + 48 + m15) * N_ + hi * 16;

    v8f acc0 = {0.f,0.f,0.f,0.f,0.f,0.f,0.f,0.f};
    v8f acc1 = acc0, acc2 = acc0, acc3 = acc0;

    for (int kk = 0; kk < N_; kk += 32) {
        // A fragment: 16 f32 per lane -> 16 f16 in contiguous K order (k = 16*hi + s)
        float4 f0 = *(const float4*)(ag0 + kk);
        float4 f1 = *(const float4*)(ag0 + kk + 4);
        float4 f2 = *(const float4*)(ag1 + kk);
        float4 f3 = *(const float4*)(ag1 + kk + 4);
        v16h a;
        a[0]  = (_Float16)f0.x; a[1]  = (_Float16)f0.y;
        a[2]  = (_Float16)f0.z; a[3]  = (_Float16)f0.w;
        a[4]  = (_Float16)f1.x; a[5]  = (_Float16)f1.y;
        a[6]  = (_Float16)f1.z; a[7]  = (_Float16)f1.w;
        a[8]  = (_Float16)f2.x; a[9]  = (_Float16)f2.y;
        a[10] = (_Float16)f2.z; a[11] = (_Float16)f2.w;
        a[12] = (_Float16)f3.x; a[13] = (_Float16)f3.y;
        a[14] = (_Float16)f3.z; a[15] = (_Float16)f3.w;

        union { v16h v; v8h h[2]; } b0u, b1u, b2u, b3u;
        const v8h* r0 = (const v8h*)(bp0 + kk);
        const v8h* r1 = (const v8h*)(bp1 + kk);
        const v8h* r2 = (const v8h*)(bp2 + kk);
        const v8h* r3 = (const v8h*)(bp3 + kk);
        b0u.h[0] = r0[0]; b0u.h[1] = r0[1];
        b1u.h[0] = r1[0]; b1u.h[1] = r1[1];
        b2u.h[0] = r2[0]; b2u.h[1] = r2[1];
        b3u.h[0] = r3[0]; b3u.h[1] = r3[1];

        acc0 = __builtin_amdgcn_wmma_f32_16x16x32_f16(false, a, false, b0u.v,
                                                      (short)0, acc0, false, false);
        acc1 = __builtin_amdgcn_wmma_f32_16x16x32_f16(false, a, false, b1u.v,
                                                      (short)0, acc1, false, false);
        acc2 = __builtin_amdgcn_wmma_f32_16x16x32_f16(false, a, false, b2u.v,
                                                      (short)0, acc2, false, false);
        acc3 = __builtin_amdgcn_wmma_f32_16x16x32_f16(false, a, false, b3u.v,
                                                      (short)0, acc3, false, false);

        if (kk + 32 < N_) __builtin_prefetch(ag0 + kk + 32, 0, 1);
    }

    // C/D layout: lanes 0-15 -> M=r, lanes 16-31 -> M=8+r; N = c*16 + (lane&15)
    const int mbase = rt * 16 + hi * 8;
    for (int r = 0; r < 8; ++r) {
        size_t o = ((size_t)b * K1_ + mbase + r) * NH_ + m15;
        Out[o +  0] = sigm(acc0[r]);
        Out[o + 16] = sigm(acc1[r]);
        Out[o + 32] = sigm(acc2[r]);
        Out[o + 48] = sigm(acc3[r]);
    }
}

// ---------------- 9) per-row bilinear pair (stages 1 and 2) ----------------
__global__ void bilinear_kernel(const float* __restrict__ xp, const float* __restrict__ xn,
                                const float* __restrict__ y1, const float* __restrict__ y2,
                                const float* __restrict__ W, const float* __restrict__ bias,
                                const float* __restrict__ mixp, float* __restrict__ out, int K) {
    __shared__ float sW[NH_ * NH_];
    for (int i = threadIdx.x; i < NH_ * NH_; i += blockDim.x) sW[i] = W[i];
    __syncthreads();
    int gw = (int)((blockIdx.x * blockDim.x + threadIdx.x) >> 5);  // row id in [0, B*K)
    int lane = threadIdx.x & 31;
    int b = gw / K, j = gw - b * K;
    size_t base = (size_t)gw * NH_;
    float y1a = y1[base + lane], y1b = y1[base + lane + 32];
    float y2a = y2[base + lane], y2b = y2[base + lane + 32];
    int i0 = lane, i1 = lane + 32;
    float u1_0 = 0.f, u1_1 = 0.f, u2_0 = 0.f, u2_1 = 0.f;
    for (int q = 0; q < 32; ++q) {
        float w00 = sW[i0 * NH_ + q], w01 = sW[i0 * NH_ + 32 + q];
        float w10 = sW[i1 * NH_ + q], w11 = sW[i1 * NH_ + 32 + q];
        float y1q = __shfl(y1a, q, 32), y1qb = __shfl(y1b, q, 32);
        float y2q = __shfl(y2a, q, 32), y2qb = __shfl(y2b, q, 32);
        u1_0 += w00 * y1q + w01 * y1qb;
        u1_1 += w10 * y1q + w11 * y1qb;
        u2_0 += w00 * y2q + w01 * y2qb;
        u2_1 += w10 * y2q + w11 * y2qb;
    }
    float xpa = xp[base + lane], xpb = xp[base + lane + 32];
    float xna = xn[base + lane], xnb = xn[base + lane + 32];
    float d1p = xpa * u1_0 + xpb * u1_1;
    float d2p = xpa * u2_0 + xpb * u2_1;
    float d1n = xna * u1_0 + xnb * u1_1;
    float d2n = xna * u2_0 + xnb * u2_1;
    wave_reduce4(d1p, d2p, d1n, d2n);
    if (lane == 0) {
        float bb = bias[0], mx = mixp[0];
        out[(size_t)b * (2 * K) + j]     = mx * (d1p + bb) + (1.0f - mx) * (d2p + bb);
        out[(size_t)b * (2 * K) + K + j] = mx * (d1n + bb) + (1.0f - mx) * (d2n + bb);
    }
}

// ---------------- 10) stage-2 gathers ----------------
__global__ void gather2_kernel(const float* __restrict__ selp, const float* __restrict__ seln,
                               const float* __restrict__ p1, const float* __restrict__ p2,
                               const int* __restrict__ idx, const int* __restrict__ idx_sub,
                               float* __restrict__ selps, float* __restrict__ selns,
                               float* __restrict__ l1s, float* __restrict__ l2s) {
    size_t t = (size_t)blockIdx.x * 256 + threadIdx.x;    // B*K2*NH
    int h = (int)(t & 63);
    size_t bj = t >> 6;
    int j = (int)(bj & (K2_ - 1));
    int b = (int)(bj >> 9);
    int r = idx_sub[b * K2_ + j];           // index into sel arrays (0..K1-1)
    int rn = idx[b * K1_ + r];              // original node index
    size_t ssrc = ((size_t)b * K1_ + r) * NH_ + h;
    selps[t] = selp[ssrc];
    selns[t] = seln[ssrc];
    size_t psrc = ((size_t)b * N_ + rn) * NH_ + h;
    l1s[t] = sigm(p1[psrc]);
    l2s[t] = sigm(p2[psrc]);
}

// =====================================================================
extern "C" void kernel_launch(void* const* d_in, const int* in_sizes, int n_in,
                              void* d_out, int out_size, void* d_ws, size_t ws_size,
                              hipStream_t stream) {
    (void)in_sizes; (void)n_in; (void)out_size; (void)ws_size;
    const float* adj   = (const float*)d_in[0];
    const float* diff  = (const float*)d_in[1];
    const float* pos1  = (const float*)d_in[2];
    const float* pos2  = (const float*)d_in[3];
    const float* neg1  = (const float*)d_in[4];
    const float* neg2  = (const float*)d_in[5];
    const float* W0    = (const float*)d_in[6];
    const float* b0    = (const float*)d_in[7];
    const float* W1    = (const float*)d_in[8];
    const float* b1    = (const float*)d_in[9];
    const float* W2    = (const float*)d_in[10];
    const float* b2    = (const float*)d_in[11];
    const float* alpha = (const float*)d_in[12];
    const float* beta  = (const float*)d_in[13];
    const float* lamda = (const float*)d_in[14];

    const size_t BNH = (size_t)B_ * N_ * NH_;

    char* ws = (char*)d_ws;
    size_t off = 0;
    auto carve = [&](size_t bytes) -> void* {
        void* p = ws + off;
        off = (off + bytes + 255) & ~(size_t)255;
        return p;
    };
    float*    p1    = (float*)carve(BNH * 4);
    float*    p2    = (float*)carve(BNH * 4);
    float*    fpos  = (float*)carve(BNH * 4);
    float*    fneg  = (float*)carve(BNH * 4);
    _Float16* p1T   = (_Float16*)carve(BNH * 2);
    _Float16* p2T   = (_Float16*)carve(BNH * 2);
    float*    g1    = (float*)carve((size_t)B_ * NH_ * 4);
    float*    g2    = (float*)carve((size_t)B_ * NH_ * 4);
    float*    Wg1   = (float*)carve((size_t)B_ * NH_ * 4);
    float*    Wg2   = (float*)carve((size_t)B_ * NH_ * 4);
    int*      idx   = (int*)carve((size_t)B_ * K1_ * 4);
    int*      idxs  = (int*)carve((size_t)B_ * K2_ * 4);
    float*    selp  = (float*)carve((size_t)B_ * K1_ * NH_ * 4);
    float*    seln  = (float*)carve((size_t)B_ * K1_ * NH_ * 4);
    float*    l1sel = (float*)carve((size_t)B_ * K1_ * NH_ * 4);
    float*    l2sel = (float*)carve((size_t)B_ * K1_ * NH_ * 4);
    float*    selps = (float*)carve((size_t)B_ * K2_ * NH_ * 4);
    float*    selns = (float*)carve((size_t)B_ * K2_ * NH_ * 4);
    float*    l1s   = (float*)carve((size_t)B_ * K2_ * NH_ * 4);
    float*    l2s   = (float*)carve((size_t)B_ * K2_ * NH_ * 4);

    float* out0 = (float*)d_out;                 // [B, 2N]
    float* out1 = out0 + (size_t)B_ * 2 * N_;    // [B, 2*K1]
    float* out2 = out1 + (size_t)B_ * 2 * K1_;   // [B, 2*K2]

    // 1) dropout + fusion + transposed f16
    prep_kernel<<<(int)(BNH / 256), 256, 0, stream>>>(pos1, pos2, neg1, neg2,
                                                      p1, p2, fpos, fneg, p1T, p2T);
    // 2) readouts
    mean_sig_kernel<<<B_ * NH_, 256, 0, stream>>>(p1, p2, g1, g2);
    // 3) W0 @ g
    wg_kernel<<<B_, NH_, 0, stream>>>(W0, g1, g2, Wg1, Wg2);
    // 4) stage-0 logits -> out0
    logits0_kernel<<<(B_ * N_) / 8, 256, 0, stream>>>(fpos, fneg, Wg1, Wg2, b0, alpha, out0);
    // 5) top-k stage 1 (M=2048 -> K1=1024) + cross-batch column sort
    topk_kernel<<<B_, 1024, N_ * 8, stream>>>(out0, N_, K1_, idx);
    colsort_kernel<<<(K1_ + 255) / 256, 256, 0, stream>>>(idx, K1_);
    // 6) gather fus by idx
    gather1_kernel<<<(int)((size_t)B_ * K1_ * NH_ / 256), 256, 0, stream>>>(fpos, fneg, idx,
                                                                            selp, seln);
    // 7) heavy WMMA GEMM on selected rows: l1 = sig(adj@p1), l2 = sig(diff@p2)
    //    grid: 8 row-tile groups (8 waves each) x B x {adj,diff}
    gemm_sigmoid_wmma<<<dim3(K1_ / 16 / 8, B_, 2), 256, 0, stream>>>(adj, diff, p1T, p2T, idx,
                                                                     l1sel, l2sel);
    // 8) stage-1 bilinears -> out1
    bilinear_kernel<<<(B_ * K1_) / 8, 256, 0, stream>>>(selp, seln, l1sel, l2sel,
                                                        W1, b1, beta, out1, K1_);
    // 9) top-k stage 2 (M=1024 -> K2=512) + column sort
    topk_kernel<<<B_, 512, K1_ * 8, stream>>>(out1, K1_, K2_, idxs);
    colsort_kernel<<<(K2_ + 255) / 256, 256, 0, stream>>>(idxs, K2_);
    // 10) stage-2 gathers
    gather2_kernel<<<(int)((size_t)B_ * K2_ * NH_ / 256), 256, 0, stream>>>(
        selp, seln, p1, p2, idx, idxs, selps, selns, l1s, l2s);
    // 11) stage-2 bilinears -> out2
    bilinear_kernel<<<(B_ * K2_) / 8, 256, 0, stream>>>(selps, selns, l1s, l2s,
                                                        W2, b2, lamda, out2, K2_);
}
